// quadraS_26637387170454
// MI455X (gfx1250) — matrix operand order repaired and checked by reading, not compiled, not run on hardware
//
#include <hip/hip_runtime.h>

// ---------------------------------------------------------------------------
// Types for CDNA5 WMMA (wave32, v_wmma_f32_16x16x32_bf16)
// ---------------------------------------------------------------------------
typedef __attribute__((ext_vector_type(16))) __bf16 v16bf;
typedef __attribute__((ext_vector_type(8)))  float  v8f;
typedef __attribute__((ext_vector_type(4)))  float  v4f;
typedef __attribute__((ext_vector_type(4)))  __bf16 v4bf;

typedef __attribute__((ext_vector_type(4))) unsigned int u32x4;
typedef __attribute__((ext_vector_type(8))) int          i32x8;
typedef __attribute__((ext_vector_type(4))) int          i32x4;

#if defined(__gfx1250__) && __has_builtin(__builtin_amdgcn_tensor_load_to_lds)
#define USE_TDM 1
#endif

__device__ __forceinline__ v8f wmma_bf16(v16bf a, v16bf b, v8f c) {
  // (neg_a, A, neg_b, B, c_mod, C, reuse_a, reuse_b)
  return __builtin_amdgcn_wmma_f32_16x16x32_bf16(false, a, false, b, (short)0, c,
                                                 false, false);
}

// A fragment: 16x32 bf16 tile, row-major source. lane&15 -> row, lane>>4 -> K half.
__device__ __forceinline__ v16bf ldA(const __bf16* __restrict__ base, int ld,
                                     int r0, int k0, int lane) {
  return *(const v16bf*)(base + (size_t)(r0 + (lane & 15)) * ld + k0 +
                         ((lane >> 4) << 4));
}
// B fragment: 32x16 bf16 tile, row-major source. lane&15 -> N col, lane>>4 -> K half.
__device__ __forceinline__ v16bf ldBf(const __bf16* __restrict__ base, int ld,
                                      int k0, int n0, int lane) {
  v16bf b;
#pragma unroll
  for (int s = 0; s < 16; ++s)
    b[s] = base[(size_t)(k0 + ((lane >> 4) << 4) + s) * ld + n0 + (lane & 15)];
  return b;
}

#ifdef USE_TDM
// ---------------------------------------------------------------------------
// Issue one TDM 2-D tile load (bf16 elements) global -> LDS.
// D# layout per CDNA5 ISA:
//   group0: [1:0]=count=1 | lds_addr | global_addr[56:0] | type=2
//   group1: data_size=1(2B), tensor_dim0/1, tile_dim0/1, tensor_dim0_stride
//   groups 2/3: zero (tile_dim2..4 = 0 -> unused)
// ---------------------------------------------------------------------------
__device__ __forceinline__ void tdm_load_tile_bf16(
    unsigned lds_addr, const void* gtile, unsigned tensor_w, unsigned tensor_h,
    unsigned tile_w, unsigned tile_h, unsigned long long stride_elems) {
  unsigned long long ga = (unsigned long long)gtile;
  u32x4 g0;
  g0[0] = 1u;  // count=1, user descriptor, no gather
  g0[1] = lds_addr;
  g0[2] = (unsigned)(ga & 0xFFFFFFFFu);
  g0[3] = ((unsigned)((ga >> 32) & 0x01FFFFFFu)) | (2u << 30);  // type=2
  i32x8 g1;
  g1[0] = (int)(1u << 16);                                   // data_size = 2B
  g1[1] = (int)((tensor_w & 0xFFFFu) << 16);                 // tdim0[15:0]
  g1[2] = (int)(((tensor_w >> 16) & 0xFFFFu) |               // tdim0[31:16]
                ((tensor_h & 0xFFFFu) << 16));               // tdim1[15:0]
  g1[3] = (int)(((tensor_h >> 16) & 0xFFFFu) |               // tdim1[31:16]
                ((tile_w & 0xFFFFu) << 16));                 // tile_dim0
  g1[4] = (int)(tile_h & 0xFFFFu);                           // tile_dim1 (dim2=0)
  g1[5] = (int)(unsigned)(stride_elems & 0xFFFFFFFFull);     // dim0_stride lo32
  g1[6] = (int)(unsigned)((stride_elems >> 32) & 0xFFFFull); // dim0_stride hi16
  g1[7] = 0;                                                 // dim1_stride (unused)
  i32x4 z4 = {0, 0, 0, 0};
#if __clang_major__ >= 23
  i32x8 z8 = {0, 0, 0, 0, 0, 0, 0, 0};
  __builtin_amdgcn_tensor_load_to_lds(g0, g1, z4, z4, z8, 0);
#else
  __builtin_amdgcn_tensor_load_to_lds(g0, g1, z4, z4, 0);
#endif
}

__device__ __forceinline__ unsigned lds_byte_addr(const void* p) {
  // generic LDS address: addr[31:0] is the LDS byte offset (ISA §10.2)
  return (unsigned)(unsigned long long)p;
}
#endif  // USE_TDM

// ---------------------------------------------------------------------------
// fp32 -> bf16 convert (vectorized x4)
// ---------------------------------------------------------------------------
__global__ void cvt_kernel(const float* __restrict__ src, __bf16* __restrict__ dst,
                           int n) {
  int i = (blockIdx.x * blockDim.x + threadIdx.x) * 4;
  if (i < n) {
    v4f v = *(const v4f*)(src + i);
    v4bf o;
    o.x = (__bf16)v.x; o.y = (__bf16)v.y; o.z = (__bf16)v.z; o.w = (__bf16)v.w;
    *(v4bf*)(dst + i) = o;
  }
}

// fp32 [R,C] -> bf16 transposed [C,R], LDS-tiled so both sides are coalesced
__global__ void transpose_cvt_kernel(const float* __restrict__ src,
                                     __bf16* __restrict__ dst, int R, int C) {
  __shared__ __bf16 tile[32][33];
  int c = blockIdx.x * 32 + threadIdx.x;
  int r = blockIdx.y * 32 + threadIdx.y;
  if (r < R && c < C) tile[threadIdx.y][threadIdx.x] = (__bf16)src[(size_t)r * C + c];
  __syncthreads();
  int oc = blockIdx.y * 32 + threadIdx.x;    // out col = src row
  int orow = blockIdx.x * 32 + threadIdx.y;  // out row = src col
  if (orow < C && oc < R) dst[(size_t)orow * R + oc] = tile[threadIdx.x][threadIdx.y];
}

// ---------------------------------------------------------------------------
// bias[r] = row(mat[r]) . other      (one wave32 per row, fixed-order shfl tree)
// ---------------------------------------------------------------------------
__global__ void dot_other_kernel(const float* __restrict__ mat,
                                 const float* __restrict__ other,
                                 float* __restrict__ bias, int rows, int D) {
  int w = (blockIdx.x * blockDim.x + threadIdx.x) >> 5;
  int lane = threadIdx.x & 31;
  if (w >= rows) return;
  const float* rp = mat + (size_t)w * D;
  float s = 0.f;
  for (int k = lane; k < D; k += 32) s += rp[k] * other[k];
  s += __shfl_xor(s, 16, 32);
  s += __shfl_xor(s, 8, 32);
  s += __shfl_xor(s, 4, 32);
  s += __shfl_xor(s, 2, 32);
  s += __shfl_xor(s, 1, 32);
  if (lane == 0) bias[w] = s;
}

// ---------------------------------------------------------------------------
// Quadratic form partials: part[row, strip] = sum over a 256-col strip of
// (A @ W)[row, n] * A[row, n].  Wave32 per (16-row block, strip).
// ---------------------------------------------------------------------------
#define QSTRIP 256
__global__ void quad_kernel(const __bf16* __restrict__ A,
                            const __bf16* __restrict__ W,
                            float* __restrict__ part, int rows, int D) {
  int lane = threadIdx.x & 31;
  int wid = blockIdx.x * (blockDim.x >> 5) + (threadIdx.x >> 5);
  int strips = D / QSTRIP;
  int r0 = (wid / strips) * 16;
  int strip = wid % strips;
  if (r0 >= rows) return;
  int n_base = strip * QSTRIP;

  float s[8];
#pragma unroll
  for (int r = 0; r < 8; ++r) s[r] = 0.f;

  for (int n0 = n_base; n0 < n_base + QSTRIP; n0 += 16) {
    v8f acc = {};
    for (int k0 = 0; k0 < D; k0 += 32) {
      v16bf af = ldA(A, D, r0, k0, lane);
      v16bf bf_ = ldBf(W, D, k0, n0, lane);
      acc = wmma_bf16(af, bf_, acc);
    }
    int col = n0 + (lane & 15);
    int rbase = r0 + ((lane >> 4) << 3);
#pragma unroll
    for (int r = 0; r < 8; ++r)
      s[r] += acc[r] * (float)A[(size_t)(rbase + r) * D + col];
  }
#pragma unroll
  for (int r = 0; r < 8; ++r) {
    float v = s[r];
    v += __shfl_xor(v, 1, 32);
    v += __shfl_xor(v, 2, 32);
    v += __shfl_xor(v, 4, 32);
    v += __shfl_xor(v, 8, 32);
    if ((lane & 15) == 0) {
      int row = r0 + ((lane >> 4) << 3) + r;
      part[(size_t)row * strips + strip] = v;
    }
  }
}

__global__ void finalize_bias_kernel(float* __restrict__ bias,
                                     const float* __restrict__ part, int rows,
                                     int strips) {
  int r = blockIdx.x * blockDim.x + threadIdx.x;
  if (r >= rows) return;
  float s = bias[r];
  for (int i = 0; i < strips; ++i) s += part[(size_t)r * strips + i];
  bias[r] = s;
}

// ---------------------------------------------------------------------------
// BM[k,j] = sum_l between[k,l] * modelsT[l,j]   (bf16 out, wave = 16x16 tile)
// ---------------------------------------------------------------------------
__global__ void bm_kernel(const __bf16* __restrict__ Bmat,
                          const __bf16* __restrict__ MT,
                          __bf16* __restrict__ BM, int M, int N, int K) {
  int lane = threadIdx.x & 31;
  int wid = blockIdx.x * (blockDim.x >> 5) + (threadIdx.x >> 5);
  int ntiles = N / 16;
  int m0 = (wid / ntiles) * 16;
  int n0 = (wid % ntiles) * 16;
  if (m0 >= M) return;
  v8f acc = {};
  for (int k0 = 0; k0 < K; k0 += 32) {
    v16bf a = ldA(Bmat, K, m0, k0, lane);
    v16bf b = ldBf(MT, N, k0, n0, lane);
    acc = wmma_bf16(a, b, acc);
  }
  int col = n0 + (lane & 15);
  int rbase = m0 + ((lane >> 4) << 3);
#pragma unroll
  for (int r = 0; r < 8; ++r)
    BM[(size_t)(rbase + r) * N + col] = (__bf16)acc[r];
}

// ---------------------------------------------------------------------------
// Main fused GEMM: out = 2 * (xb @ BM) + rowbias[i] + colbias[j]
// 128x128 WG tile, 8 waves each 32x64 (2x4 WMMA tiles), double-buffered LDS.
// Tile staging via TDM (tensor_load_to_lds + s_wait_tensorcnt) when available.
// ---------------------------------------------------------------------------
#define TM 128
#define TN 128
#define TK 32

__global__ __launch_bounds__(256) void main_gemm_kernel(
    const __bf16* __restrict__ A, const __bf16* __restrict__ Bm,
    const float* __restrict__ rowbias, const float* __restrict__ colbias,
    float* __restrict__ out, int M, int N, int K) {
  __shared__ __bf16 lA[2][TM * TK];  // 2 x 8 KB
  __shared__ __bf16 lB[2][TK * TN];  // 2 x 8 KB

  int tid = threadIdx.x;
  int lane = tid & 31;
  int w = tid >> 5;            // 0..7
  int wm = (w >> 1) * 32;      // wave M offset in tile
  int wn = (w & 1) * 64;       // wave N offset in tile
  int mBase = blockIdx.y * TM;
  int nBase = blockIdx.x * TN;

  v8f zero = {};
  v8f acc[2][4];
#pragma unroll
  for (int mi = 0; mi < 2; ++mi)
#pragma unroll
    for (int ni = 0; ni < 4; ++ni) acc[mi][ni] = zero;

  int NKT = K / TK;

#ifdef USE_TDM
  // -------- TDM-staged pipeline: wave 0 drives the Tensor Data Mover --------
  if (w == 0) {
    tdm_load_tile_bf16(lds_byte_addr(&lA[0][0]),
                       A + (size_t)mBase * K, (unsigned)K, (unsigned)M,
                       TK, TM, (unsigned long long)K);
    tdm_load_tile_bf16(lds_byte_addr(&lB[0][0]),
                       Bm + nBase, (unsigned)N, (unsigned)K,
                       TN, TK, (unsigned long long)N);
  }
  for (int kt = 0; kt < NKT; ++kt) {
    int cur = kt & 1, nxt = cur ^ 1;
    if (w == 0) {
      if (kt + 1 < NKT) {
        tdm_load_tile_bf16(lds_byte_addr(&lA[nxt][0]),
                           A + (size_t)mBase * K + (kt + 1) * TK,
                           (unsigned)K, (unsigned)M, TK, TM,
                           (unsigned long long)K);
        tdm_load_tile_bf16(lds_byte_addr(&lB[nxt][0]),
                           Bm + (size_t)(kt + 1) * TK * N + nBase,
                           (unsigned)N, (unsigned)K, TN, TK,
                           (unsigned long long)N);
        // 2 newest ops may stay in flight; the older pair (buffer `cur`) done
        __builtin_amdgcn_s_wait_tensorcnt(2);
      } else {
        __builtin_amdgcn_s_wait_tensorcnt(0);
      }
    }
    __syncthreads();  // buffer `cur` visible to all waves

    v16bf af[2], bf_[4];
#pragma unroll
    for (int mi = 0; mi < 2; ++mi)
      af[mi] = *(const v16bf*)&lA[cur][(wm + mi * 16 + (lane & 15)) * TK +
                                       ((lane >> 4) << 4)];
#pragma unroll
    for (int ni = 0; ni < 4; ++ni)
#pragma unroll
      for (int s = 0; s < 16; ++s)
        bf_[ni][s] = lB[cur][(((lane >> 4) << 4) + s) * TN + wn + ni * 16 + (lane & 15)];

#pragma unroll
    for (int mi = 0; mi < 2; ++mi)
#pragma unroll
      for (int ni = 0; ni < 4; ++ni)
        acc[mi][ni] = wmma_bf16(af[mi], bf_[ni], acc[mi][ni]);

    __syncthreads();  // everyone done with `cur` before TDM re-targets it
  }
#else
  // -------- fallback: register-staged double-buffered pipeline --------
  int ar = tid >> 1, ah = (tid & 1) << 4;   // A: row, k-offset
  int br = tid >> 3, bs = (tid & 7) << 4;   // B: k-row, n-offset

  const __bf16* gA = A + (size_t)(mBase + ar) * K + ah;
  const __bf16* gB = Bm + (size_t)br * N + nBase + bs;

  v16bf ra = *(const v16bf*)(gA);
  v16bf rb = *(const v16bf*)(gB);
  *(v16bf*)&lA[0][ar * TK + ah] = ra;
  *(v16bf*)&lB[0][br * TN + bs] = rb;
  __syncthreads();

  for (int kt = 0; kt < NKT; ++kt) {
    int cur = kt & 1, nxt = cur ^ 1;
    if (kt + 1 < NKT) {
      ra = *(const v16bf*)(gA + (size_t)(kt + 1) * TK);
      rb = *(const v16bf*)(gB + (size_t)(kt + 1) * TK * N);
    }
    if (kt + 2 < NKT) {
      __builtin_prefetch(gA + (size_t)(kt + 2) * TK, 0, 1);
      __builtin_prefetch(gB + (size_t)(kt + 2) * TK * N, 0, 1);
    }

    v16bf af[2], bf_[4];
#pragma unroll
    for (int mi = 0; mi < 2; ++mi)
      af[mi] = *(const v16bf*)&lA[cur][(wm + mi * 16 + (lane & 15)) * TK +
                                       ((lane >> 4) << 4)];
#pragma unroll
    for (int ni = 0; ni < 4; ++ni)
#pragma unroll
      for (int s = 0; s < 16; ++s)
        bf_[ni][s] = lB[cur][(((lane >> 4) << 4) + s) * TN + wn + ni * 16 + (lane & 15)];

#pragma unroll
    for (int mi = 0; mi < 2; ++mi)
#pragma unroll
      for (int ni = 0; ni < 4; ++ni)
        acc[mi][ni] = wmma_bf16(af[mi], bf_[ni], acc[mi][ni]);

    if (kt + 1 < NKT) {
      *(v16bf*)&lA[nxt][ar * TK + ah] = ra;
      *(v16bf*)&lB[nxt][br * TN + bs] = rb;
    }
    __syncthreads();
  }
#endif

  // fused epilogue: 2*acc + rowbias + colbias
#pragma unroll
  for (int mi = 0; mi < 2; ++mi) {
    int rbase = mBase + wm + mi * 16 + ((lane >> 4) << 3);
#pragma unroll
    for (int ni = 0; ni < 4; ++ni) {
      int col = nBase + wn + ni * 16 + (lane & 15);
      float cb = colbias[col];
#pragma unroll
      for (int r = 0; r < 8; ++r)
        out[(size_t)(rbase + r) * N + col] =
            2.0f * acc[mi][ni][r] + rowbias[rbase + r] + cb;
    }
  }
}

// ---------------------------------------------------------------------------
// Host launcher
// ---------------------------------------------------------------------------
extern "C" void kernel_launch(void* const* d_in, const int* in_sizes, int n_in,
                              void* d_out, int out_size, void* d_ws, size_t ws_size,
                              hipStream_t stream) {
  constexpr int Bn = 8192, Dn = 1024, On = 4096;

  const float* x       = (const float*)d_in[0];
  const float* within  = (const float*)d_in[1];
  const float* between = (const float*)d_in[2];
  const float* other   = (const float*)d_in[3];
  const float* models  = (const float*)d_in[4];
  float* out = (float*)d_out;

  char* ws = (char*)d_ws;
  size_t off = 0;
  auto carve = [&](size_t bytes) {
    void* p = ws + off;
    off += (bytes + 255) & ~(size_t)255;
    return p;
  };
  __bf16* xb  = (__bf16*)carve((size_t)Bn * Dn * 2);  // x in bf16
  __bf16* mb  = (__bf16*)carve((size_t)On * Dn * 2);  // models in bf16
  __bf16* wb  = (__bf16*)carve((size_t)Dn * Dn * 2);  // within in bf16
  __bf16* bb  = (__bf16*)carve((size_t)Dn * Dn * 2);  // between in bf16
  __bf16* mt  = (__bf16*)carve((size_t)Dn * On * 2);  // models^T in bf16
  __bf16* bmb = (__bf16*)carve((size_t)Dn * On * 2);  // BM = between @ models^T
  float* rowbias = (float*)carve((size_t)Bn * 4);
  float* colbias = (float*)carve((size_t)On * 4);
  const int strips = Dn / QSTRIP;  // 4
  float* partX = (float*)carve((size_t)Bn * strips * 4);
  float* partM = (float*)carve((size_t)On * strips * 4);

  // 1) precision conversions
  cvt_kernel<<<(Bn * Dn / 4 + 255) / 256, 256, 0, stream>>>(x, xb, Bn * Dn);
  cvt_kernel<<<(On * Dn / 4 + 255) / 256, 256, 0, stream>>>(models, mb, On * Dn);
  cvt_kernel<<<(Dn * Dn / 4 + 255) / 256, 256, 0, stream>>>(within, wb, Dn * Dn);
  cvt_kernel<<<(Dn * Dn / 4 + 255) / 256, 256, 0, stream>>>(between, bb, Dn * Dn);
  transpose_cvt_kernel<<<dim3(Dn / 32, On / 32), dim3(32, 32), 0, stream>>>(
      models, mt, On, Dn);

  // 2) biases: dot with `other` (fp32), then add quadratic-form partials
  dot_other_kernel<<<Bn / 8, 256, 0, stream>>>(x, other, rowbias, Bn, Dn);
  dot_other_kernel<<<On / 8, 256, 0, stream>>>(models, other, colbias, On, Dn);

  quad_kernel<<<(Bn / 16) * strips / 8, 256, 0, stream>>>(xb, wb, partX, Bn, Dn);
  quad_kernel<<<(On / 16) * strips / 8, 256, 0, stream>>>(mb, wb, partM, On, Dn);
  finalize_bias_kernel<<<Bn / 256, 256, 0, stream>>>(rowbias, partX, Bn, strips);
  finalize_bias_kernel<<<On / 256, 256, 0, stream>>>(colbias, partM, On, strips);

  // 3) BM = between @ models^T  [Dn x On], bf16
  bm_kernel<<<(Dn / 16) * (On / 16) / 8, 256, 0, stream>>>(bb, mt, bmb, Dn, On, Dn);

  // 4) out = 2 * xb @ BM + rowbias + colbias   [Bn x On], fp32
  main_gemm_kernel<<<dim3(On / TN, Bn / TM), 256, 0, stream>>>(
      xb, bmb, rowbias, colbias, out, Bn, On, Dn);
}
